// GatedGCNModel_77713138253857
// MI455X (gfx1250) — matrix operand order repaired and checked by reading, not compiled, not run on hardware
//
#include <hip/hip_runtime.h>
#include <stdint.h>

#define IN_DIM 64
#define ENC_H  256
#define HDIM   128
#define NLAYERS 4
#define EPSN   1e-5f

typedef __attribute__((ext_vector_type(16))) __bf16    v16bf;
typedef __attribute__((ext_vector_type(8)))  float     v8f;
typedef __attribute__((ext_vector_type(4)))  uint32_t  u32x4;
typedef __attribute__((ext_vector_type(8)))  int32_t   i32x8;
typedef __attribute__((ext_vector_type(4)))  int32_t   i32x4;

#ifndef __has_builtin
#define __has_builtin(x) 0
#endif
#if __has_builtin(__builtin_amdgcn_tensor_load_to_lds)
#define HAVE_TDM 1
#else
#define HAVE_TDM 0
#endif

__device__ __forceinline__ uint16_t f2bf(float f) {
    uint32_t u = __float_as_uint(f);
    uint32_t r = u + 0x7FFFu + ((u >> 16) & 1u);   // round-to-nearest-even
    return (uint16_t)(r >> 16);
}

// ---------------------------------------------------------------- utilities
__global__ void k_zero(float* p, int n) {
    int i = blockIdx.x * blockDim.x + threadIdx.x;
    int s = gridDim.x * blockDim.x;
    for (; i < n; i += s) p[i] = 0.f;
}

__global__ void k_deg(const int* __restrict__ row, float* __restrict__ deg, int E) {
    int e = blockIdx.x * blockDim.x + threadIdx.x;
    if (e < E) atomicAdd(&deg[row[e]], 1.0f);
}

__global__ void k_cvt_bf16(const float* __restrict__ s, uint16_t* __restrict__ d, int n) {
    int i = blockIdx.x * blockDim.x + threadIdx.x;
    if (i < n) d[i] = f2bf(s[i]);
}

// -------------------------------------------- weight pre-pack (B fragments)
// Pack W[K x Ncsub] (f32 row-major) into WMMA B-fragment order:
//   dst[(((slab*(K/32) + kb)*32 + lane)*16 + j]
//   slab = c/16, nn = c%16, kb = k/32, kk = k%32, lane = nn + 16*(kk/16), j = kk%16
__global__ void k_pack_w(const float* __restrict__ w, uint16_t* __restrict__ p,
                         int K, int Ncsub) {
    int i = blockIdx.x * blockDim.x + threadIdx.x;
    if (i >= K * Ncsub) return;
    int k = i / Ncsub, c = i - k * Ncsub;
    int slab = c >> 4, nn = c & 15;
    int kb = k >> 5,  kk = k & 31;
    int lane = nn + ((kk >> 4) << 4);
    int j = kk & 15;
    p[(((size_t)slab * (K >> 5) + kb) * 32 + lane) * 16 + j] = f2bf(w[i]);
}

// --------------------------------------------- TDM: load rows x 32 bf16 tile
#if HAVE_TDM
__device__ __forceinline__ void tdm_load_a(uint32_t lds_off, const uint16_t* gA,
                                           int K, int rows_remain, int rows_tile) {
    uint64_t ga = (uint64_t)(uintptr_t)gA;
    u32x4 g0 = { 1u,                                    // count=1 (valid D#)
                 lds_off,                               // lds_addr
                 (uint32_t)ga,
                 ((uint32_t)(ga >> 32) & 0x1FFFFFFu) | (2u << 30) }; // type=2
    uint32_t Ku = (uint32_t)K, Ru = (uint32_t)rows_remain;
    i32x8 g1 = { (int)(1u << 16),                               // data_size = 2B
                 (int)((Ku & 0xFFFFu) << 16),                   // tensor_dim0 lo
                 (int)(((Ku >> 16) & 0xFFFFu) | ((Ru & 0xFFFFu) << 16)),
                 (int)(((Ru >> 16) & 0xFFFFu) | (32u << 16)),   // tile_dim0 = 32
                 (int)(uint32_t)rows_tile,                      // tile_dim1
                 (int)Ku,                                       // dim0_stride lo
                 0, 0 };
    i32x4 z4 = {};
#if __clang_major__ >= 23
    i32x8 z8 = {};
    __builtin_amdgcn_tensor_load_to_lds(g0, g1, z4, z4, z8, 0);
#else
    __builtin_amdgcn_tensor_load_to_lds(g0, g1, z4, z4, 0);
#endif
}
#endif

// --------------------------------------------------------- bf16 WMMA GEMM
// C[M,Nc] = A[M,K](bf16 row-major) @ Bpacked (+bias) (+relu), fp32 out.
// Block: 128 thr = 4 waves. Tile 32(M) x 64(N), K-step 32; 2 WMMA per wave
// per K-step sharing one B fragment. A tiles double-buffered in LDS via TDM
// (peeled s_wait_tensorcnt(1)/(0) pipeline). Optional fused column stats
// (sum / sum^2 atomics) for BatchNorm.
__global__ void k_gemm_bf16(const uint16_t* __restrict__ A,
                            const uint16_t* __restrict__ Bp,
                            const float* __restrict__ bias,
                            float* __restrict__ C,
                            float* __restrict__ stats,
                            int M, int K, int Nc, int relu) {
    __shared__ uint16_t ldsA[2][32 * 32];
    const int tid  = threadIdx.x;
    const int wave = tid >> 5;
    const int lane = tid & 31;
    const int m0 = blockIdx.x * 32;
    const int n0 = blockIdx.y * 64;
    const int nk = K >> 5;

    const int slab = (n0 >> 4) + wave;                          // 16-col slab id
    const uint16_t* bbase = Bp + ((size_t)slab * nk) * 512 + (size_t)lane * 16;

    v8f acc0 = {}, acc1 = {};
    const int mr = lane & 15, half = lane >> 4;

#if HAVE_TDM
    uint32_t lds0 = (uint32_t)(uintptr_t)&ldsA[0][0];
    uint32_t lds1 = (uint32_t)(uintptr_t)&ldsA[1][0];
    if (tid == 0) tdm_load_a(lds0, A + (size_t)m0 * K, K, M - m0, 32); // prologue
#endif

    auto compute = [&](int ib) {
        const uint16_t* la = &ldsA[ib & 1][0];
        union { v16bf v; uint4 q[2]; } a0, a1;
        a0.q[0] = *(const uint4*)&la[mr * 32 + half * 16];
        a0.q[1] = *(const uint4*)&la[mr * 32 + half * 16 + 8];
        a1.q[0] = *(const uint4*)&la[(16 + mr) * 32 + half * 16];
        a1.q[1] = *(const uint4*)&la[(16 + mr) * 32 + half * 16 + 8];
        union { v16bf v; uint4 q[2]; } bfr;
        const uint16_t* bp = bbase + (size_t)ib * 512;
        bfr.q[0] = *(const uint4*)bp;
        bfr.q[1] = *(const uint4*)(bp + 8);
        acc0 = __builtin_amdgcn_wmma_f32_16x16x32_bf16(
            false, a0.v, false, bfr.v, (short)0, acc0, false, false);
        acc1 = __builtin_amdgcn_wmma_f32_16x16x32_bf16(
            false, a1.v, false, bfr.v, (short)0, acc1, false, false);
    };
#if !HAVE_TDM
    auto stage = [&](int ib) {   // cooperative fallback: 8 bf16 per thread
        int e = tid * 8; int ar = e >> 5, ac = e & 31; int m = m0 + ar;
        uint4 av; av.x = av.y = av.z = av.w = 0u;
        if (m < M) av = *(const uint4*)&A[(size_t)m * K + ib * 32 + ac];
        *(uint4*)&ldsA[ib & 1][ar * 32 + ac] = av;
    };
#endif

    // steady state: issue next tile, wait for current (1 outstanding)
    int ib = 0;
    for (; ib + 1 < nk; ++ib) {
#if HAVE_TDM
        if (tid == 0)
            tdm_load_a((ib & 1) ? lds0 : lds1,
                       A + (size_t)m0 * K + (ib + 1) * 32, K, M - m0, 32);
        if (wave == 0) __builtin_amdgcn_s_wait_tensorcnt(1);
#else
        stage(ib);
#endif
        __syncthreads();
        compute(ib);
        __syncthreads();
    }
    // final iteration: drain
#if HAVE_TDM
    if (wave == 0) __builtin_amdgcn_s_wait_tensorcnt(0);
#else
    stage(ib);
#endif
    __syncthreads();
    compute(ib);

    // epilogue (C layout: VGPR r -> M=r (lanes 0-15) / M=r+8 (lanes 16-31))
    const int n  = n0 + wave * 16 + mr;
    const int mb = half * 8;
    const float bi = bias ? bias[n] : 0.f;
    float s = 0.f, s2 = 0.f;
#pragma unroll
    for (int r = 0; r < 8; ++r) {
        int m = m0 + mb + r;
        if (m < M) {
            float v = acc0[r] + bi;
            if (relu) v = fmaxf(v, 0.f);
            C[(size_t)m * Nc + n] = v;
            s += v; s2 += v * v;
        }
    }
#pragma unroll
    for (int r = 0; r < 8; ++r) {
        int m = m0 + 16 + mb + r;
        if (m < M) {
            float v = acc1[r] + bi;
            if (relu) v = fmaxf(v, 0.f);
            C[(size_t)m * Nc + n] = v;
            s += v; s2 += v * v;
        }
    }
    if (stats) {                       // fused BatchNorm column statistics
        atomicAdd(&stats[n], s);
        atomicAdd(&stats[Nc + n], s2);
    }
}

// -------------------------------------------------- BatchNorm apply (+bf16)
__global__ void k_bn_relu(const float* __restrict__ z, const float* __restrict__ st,
                          const float* __restrict__ g, const float* __restrict__ b,
                          float* __restrict__ hf, uint16_t* __restrict__ hb,
                          int rows, int cols) {
    int i = blockIdx.x * blockDim.x + threadIdx.x;
    if (i >= rows * cols) return;
    int c = i % cols;
    float inv = 1.f / (float)rows;
    float mu  = st[c] * inv;
    float var = st[cols + c] * inv - mu * mu;
    float v = (z[i] - mu) * rsqrtf(var + EPSN) * g[c] + b[c];
    v = fmaxf(v, 0.f);
    hf[i] = v;
    hb[i] = f2bf(v);
}

// ----------------------------------------- per-edge gate + message scatter
// one wave per edge. t = [a | b | hW] combined, row stride 3*HDIM.
__global__ void k_edge(const int* __restrict__ row, const int* __restrict__ col,
                       const float* __restrict__ t, const float* __restrict__ gb1,
                       const float* __restrict__ gw2, const float* __restrict__ gb2,
                       float* __restrict__ agg, int E) {
    int e = blockIdx.x * (blockDim.x >> 5) + (threadIdx.x >> 5);
    if (e >= E) return;
    int lane = threadIdx.x & 31;
    int r = row[e], c = col[e];
    const size_t TS = 3 * HDIM;
    float4 av = *(const float4*)&t[(size_t)r * TS + lane * 4];
    float4 bv = *(const float4*)&t[(size_t)c * TS + HDIM + lane * 4];
    float4 bi = *(const float4*)&gb1[lane * 4];
    float4 wv = *(const float4*)&gw2[lane * 4];
    float ux = fmaxf(av.x + bv.x + bi.x, 0.f);
    float uy = fmaxf(av.y + bv.y + bi.y, 0.f);
    float uz = fmaxf(av.z + bv.z + bi.z, 0.f);
    float uw = fmaxf(av.w + bv.w + bi.w, 0.f);
    float s = ux * wv.x + uy * wv.y + uz * wv.z + uw * wv.w;
#pragma unroll
    for (int o = 16; o >= 1; o >>= 1) s += __shfl_xor(s, o, 32);
    float gate = 1.f / (1.f + __expf(-(s + gb2[0])));
    float4 hv = *(const float4*)&t[(size_t)c * TS + 2 * HDIM + lane * 4];
    float* dst = &agg[(size_t)r * HDIM + lane * 4];
    atomicAdd(dst + 0, gate * hv.x);
    atomicAdd(dst + 1, gate * hv.y);
    atomicAdd(dst + 2, gate * hv.z);
    atomicAdd(dst + 3, gate * hv.w);
}

// ---------------------------- residual + mean-aggregate + LayerNorm + ReLU
__global__ void k_node(const float* __restrict__ h, const float* __restrict__ agg,
                       const float* __restrict__ deg,
                       const float* __restrict__ lng, const float* __restrict__ lnb,
                       float* __restrict__ hout, uint16_t* __restrict__ hbout, int n) {
    int node = blockIdx.x * (blockDim.x >> 5) + (threadIdx.x >> 5);
    if (node >= n) return;
    int lane = threadIdx.x & 31;
    float inv = 1.f / fmaxf(deg[node], 1.f);
    float4 hv = *(const float4*)&h[(size_t)node * HDIM + lane * 4];
    float4 av = *(const float4*)&agg[(size_t)node * HDIM + lane * 4];
    float4 t;
    t.x = hv.x + av.x * inv; t.y = hv.y + av.y * inv;
    t.z = hv.z + av.z * inv; t.w = hv.w + av.w * inv;
    float s  = t.x + t.y + t.z + t.w;
    float s2 = t.x * t.x + t.y * t.y + t.z * t.z + t.w * t.w;
#pragma unroll
    for (int o = 16; o >= 1; o >>= 1) {
        s  += __shfl_xor(s, o, 32);
        s2 += __shfl_xor(s2, o, 32);
    }
    float mu  = s * (1.f / HDIM);
    float var = s2 * (1.f / HDIM) - mu * mu;
    float rs  = rsqrtf(var + EPSN);
    float4 g = *(const float4*)&lng[lane * 4];
    float4 b = *(const float4*)&lnb[lane * 4];
    float4 o4;
    o4.x = fmaxf((t.x - mu) * rs * g.x + b.x, 0.f);
    o4.y = fmaxf((t.y - mu) * rs * g.y + b.y, 0.f);
    o4.z = fmaxf((t.z - mu) * rs * g.z + b.z, 0.f);
    o4.w = fmaxf((t.w - mu) * rs * g.w + b.w, 0.f);
    *(float4*)&hout[(size_t)node * HDIM + lane * 4] = o4;
    uint2 pk;
    pk.x = (uint32_t)f2bf(o4.x) | ((uint32_t)f2bf(o4.y) << 16);
    pk.y = (uint32_t)f2bf(o4.z) | ((uint32_t)f2bf(o4.w) << 16);
    *(uint2*)&hbout[(size_t)node * HDIM + lane * 4] = pk;
}

// ------------------------------------------------------------- head layer 2
__global__ void k_head2(const float* __restrict__ z, const float* __restrict__ w2,
                        const float* __restrict__ b2, float* __restrict__ out, int n) {
    int node = blockIdx.x * (blockDim.x >> 5) + (threadIdx.x >> 5);
    if (node >= n) return;
    int lane = threadIdx.x & 31;
    float2 zv = *(const float2*)&z[(size_t)node * 64 + lane * 2];
    float2 wv = *(const float2*)&w2[lane * 2];
    float s = zv.x * wv.x + zv.y * wv.y;
#pragma unroll
    for (int o = 16; o >= 1; o >>= 1) s += __shfl_xor(s, o, 32);
    if (lane == 0) out[node] = s + b2[0];
}

// ===========================================================================
extern "C" void kernel_launch(void* const* d_in, const int* in_sizes, int n_in,
                              void* d_out, int out_size, void* d_ws, size_t ws_size,
                              hipStream_t stream) {
    const float* x       = (const float*)d_in[0];
    const int*   eidx    = (const int*)d_in[1];
    const float* enc_w1  = (const float*)d_in[2];
    const float* enc_b1  = (const float*)d_in[3];
    const float* bn1_g   = (const float*)d_in[4];
    const float* bn1_b   = (const float*)d_in[5];
    const float* enc_w2  = (const float*)d_in[6];
    const float* enc_b2  = (const float*)d_in[7];
    const float* bn2_g   = (const float*)d_in[8];
    const float* bn2_b   = (const float*)d_in[9];
    const float* Wl      = (const float*)d_in[10];
    const float* gate_w1 = (const float*)d_in[11];
    const float* gate_b1 = (const float*)d_in[12];
    const float* gate_w2 = (const float*)d_in[13];
    const float* gate_b2 = (const float*)d_in[14];
    const float* ln_g    = (const float*)d_in[15];
    const float* ln_b    = (const float*)d_in[16];
    const float* head_w1 = (const float*)d_in[17];
    const float* head_b1 = (const float*)d_in[18];
    const float* head_w2 = (const float*)d_in[19];
    const float* head_b2 = (const float*)d_in[20];
    float* out = (float*)d_out;

    const int N = in_sizes[0] / IN_DIM;
    const int E = in_sizes[1] / 2;
    const int* row = eidx;
    const int* col = eidx + E;

    uintptr_t base = (uintptr_t)d_ws;
    size_t off = 0;
    auto alloc = [&](size_t bytes) -> void* {
        void* p = (void*)(base + off);
        off = (off + bytes + 255) & ~(size_t)255;
        return p;
    };
    float*    z1    = (float*)alloc((size_t)N * ENC_H * 4);     // also z2/head z
    float*    h     = (float*)alloc((size_t)N * HDIM * 4);
    float*    tcomb = (float*)alloc((size_t)N * 3 * HDIM * 4);  // [a|b|hW]
    float*    agg   = (float*)alloc((size_t)N * HDIM * 4);
    float*    deg   = (float*)alloc((size_t)N * 4);
    float*    st    = (float*)alloc(512 * 4);
    uint16_t* xb    = (uint16_t*)alloc((size_t)N * IN_DIM * 2);
    uint16_t* h1b   = (uint16_t*)alloc((size_t)N * ENC_H * 2);
    uint16_t* hb    = (uint16_t*)alloc((size_t)N * HDIM * 2);
    uint16_t* wp1   = (uint16_t*)alloc((size_t)IN_DIM * ENC_H * 2); // packed enc_w1
    uint16_t* wp2   = (uint16_t*)alloc((size_t)ENC_H * HDIM * 2);   // packed enc_w2
    uint16_t* wpL   = (uint16_t*)alloc((size_t)HDIM * 3 * HDIM * 2);// packed [gw1a|gw1b|Wl]
    uint16_t* wpH   = (uint16_t*)alloc((size_t)HDIM * 64 * 2);      // packed head_w1

    dim3 B256(256);
    auto pack = [&](const float* w, uint16_t* p, int K, int Ncsub) {
        k_pack_w<<<dim3((K * Ncsub + 255) / 256), B256, 0, stream>>>(w, p, K, Ncsub);
    };
    auto gemm = [&](const uint16_t* A, const uint16_t* Bp, const float* bias,
                    float* C, float* stats, int M, int K, int Nc, int relu) {
        k_gemm_bf16<<<dim3((M + 31) / 32, Nc / 64), dim3(128), 0, stream>>>(
            A, Bp, bias, C, stats, M, K, Nc, relu);
    };
    // one packed 128-col submatrix with K=HDIM: 8 slabs * (K/32) * 512 elems
    const size_t subOff = (size_t)8 * (HDIM >> 5) * 512;

    // degree (constant across layers)
    k_zero<<<dim3(256), B256, 0, stream>>>(deg, N);
    k_deg<<<dim3((E + 255) / 256), B256, 0, stream>>>(row, deg, E);

    // activations to bf16; weights to packed-bf16 fragment order
    k_cvt_bf16<<<dim3((N * IN_DIM + 255) / 256), B256, 0, stream>>>(x, xb, N * IN_DIM);
    pack(enc_w1, wp1, IN_DIM, ENC_H);
    pack(enc_w2, wp2, ENC_H, HDIM);
    pack(head_w1, wpH, HDIM, 64);

    // encoder layer 1: z1 = x@W1 + b1 (stats fused) ; h1 = relu(BN(z1))
    k_zero<<<dim3(2), B256, 0, stream>>>(st, 512);
    gemm(xb, wp1, enc_b1, z1, st, N, IN_DIM, ENC_H, 0);
    k_bn_relu<<<dim3((N * ENC_H + 255) / 256), B256, 0, stream>>>(
        z1, st, bn1_g, bn1_b, z1, h1b, N, ENC_H);

    // encoder layer 2: z2 = h1@W2 + b2 (stats fused) ; h = relu(BN(z2))
    k_zero<<<dim3(2), B256, 0, stream>>>(st, 512);
    gemm(h1b, wp2, enc_b2, z1, st, N, ENC_H, HDIM, 0);
    k_bn_relu<<<dim3((N * HDIM + 255) / 256), B256, 0, stream>>>(
        z1, st, bn2_g, bn2_b, h, hb, N, HDIM);

    // GatedGCN layers: one fused [N,128]@[128,384] WMMA GEMM per layer
    for (int l = 0; l < NLAYERS; ++l) {
        pack(gate_w1 + (size_t)l * 2 * HDIM * HDIM,               wpL,              HDIM, HDIM);
        pack(gate_w1 + (size_t)l * 2 * HDIM * HDIM + HDIM * HDIM, wpL + subOff,     HDIM, HDIM);
        pack(Wl      + (size_t)l * HDIM * HDIM,                   wpL + 2 * subOff, HDIM, HDIM);
        gemm(hb, wpL, nullptr, tcomb, nullptr, N, HDIM, 3 * HDIM, 0);   // [a|b|hW]
        k_zero<<<dim3(2048), B256, 0, stream>>>(agg, N * HDIM);
        k_edge<<<dim3((E + 7) / 8), B256, 0, stream>>>(
            row, col, tcomb, gate_b1 + (size_t)l * HDIM,
            gate_w2 + (size_t)l * HDIM, gate_b2 + l, agg, E);
        k_node<<<dim3((N + 7) / 8), B256, 0, stream>>>(
            h, agg, deg, ln_g + (size_t)l * HDIM, ln_b + (size_t)l * HDIM,
            h, hb, N);
    }

    // head: z = relu(h@W1 + b1) ; out = z@W2 + b2
    gemm(hb, wpH, head_b1, z1, nullptr, N, HDIM, 64, 1);
    k_head2<<<dim3((N + 7) / 8), B256, 0, stream>>>(z1, head_w2, head_b2, out, N);
}
